// DilatedMHSABlock_25512105738920
// MI455X (gfx1250) — compile-verified, matrix operand order checked by reading, not run on hardware
//
#include <hip/hip_runtime.h>
#include <hip/hip_bf16.h>

typedef __attribute__((ext_vector_type(16))) __bf16 v16bf;
typedef __attribute__((ext_vector_type(8)))  float  v8f;
typedef __attribute__((ext_vector_type(4)))  int    v4i;

#define HH     16      // heads
#define DH     64      // head dim
#define SEQ    2048
#define DMODEL 1024
#define BATCH  2
#define WIN    17      // dilated window size: j = i-16 .. i+16 step 2

#if __has_builtin(__builtin_amdgcn_global_load_async_to_lds_b128)
#define USE_ASYNC_LDS 1
#else
#define USE_ASYNC_LDS 0
#endif

#define AS1 __attribute__((address_space(1)))
#define AS3 __attribute__((address_space(3)))

// ---------------- fp32 -> bf16 (round to nearest even) ----------------
__device__ __forceinline__ unsigned short f2bf(float f) {
  unsigned u = __builtin_bit_cast(unsigned, f);
  unsigned r = u + 0x7FFFu + ((u >> 16) & 1u);
  return (unsigned short)(r >> 16);
}

__global__ __launch_bounds__(256)
void cvt_f32_bf16(const float* __restrict__ in, unsigned short* __restrict__ out, int n4) {
  int i = blockIdx.x * blockDim.x + threadIdx.x;
  if (i >= n4) return;
  float4 f = ((const float4*)in)[i];
  ushort4 o;
  o.x = f2bf(f.x); o.y = f2bf(f.y); o.z = f2bf(f.z); o.w = f2bf(f.w);
  ((ushort4*)out)[i] = o;
}

// ---------------- bf16 NT GEMM with fp32 accumulate: C = A * B^T + bias ----------------
// A: [M][Kd] bf16 row-major; Bm: [Nd][Kd] bf16 row-major (so B^T is Kd x Nd);
// C: [M][Nd] fp32.  M,Nd multiples of 128; Kd multiple of 32.
#define BM 128
#define BN 128
#define BK 32
#define LDT 40   // padded LDS row stride in ushorts (80B, 16B-aligned, bank-spread)

union Frag { v16bf v; uint4 q[2]; };

__device__ __forceinline__ void stage_chunk(const unsigned short* __restrict__ gsrc,
                                            unsigned short* lds_dst) {
#if USE_ASYNC_LDS
  // CDNA5 async DMA: global -> LDS without VGPR round trip, tracked on ASYNCcnt
  __builtin_amdgcn_global_load_async_to_lds_b128(
      (AS1 v4i*)gsrc, (AS3 v4i*)lds_dst, 0, 0);
#else
  *(uint4*)lds_dst = *(const uint4*)gsrc;
#endif
}

__device__ __forceinline__ void stage_fence() {
#if USE_ASYNC_LDS
#if __has_builtin(__builtin_amdgcn_s_wait_asynccnt)
  __builtin_amdgcn_s_wait_asynccnt(0);
#else
  asm volatile("s_wait_asynccnt 0x0" ::: "memory");
#endif
#endif
}

__global__ __launch_bounds__(256)
void gemm_nt_bias(const unsigned short* __restrict__ A,
                  const unsigned short* __restrict__ Bm,
                  const float* __restrict__ bias,
                  float* __restrict__ C,
                  int Ndim, int Kd) {
  __shared__ unsigned short As[2][BM * LDT];
  __shared__ unsigned short Bs[2][BN * LDT];

  const int tid  = threadIdx.x;
  const int lane = tid & 31;
  const int wid  = tid >> 5;
  const int wm   = wid >> 2;     // 0..1  -> 64 rows each
  const int wn   = wid & 3;      // 0..3  -> 32 cols each
  const int l16  = lane & 15;
  const int half = lane >> 4;    // 0/1: selects K sub-chunk per ISA bf16 A/B layout

  const int brow = blockIdx.y * BM;
  const int bcol = blockIdx.x * BN;
  const int T = Kd / BK;

  v8f acc[4][2] = {};

  // preload tile 0 (each thread: 2 x 16B chunks of A tile and of B tile)
  {
#pragma unroll
    for (int s = 0; s < 2; ++s) {
      int c = tid * 2 + s;          // 0..511
      int row = c >> 2, cc = c & 3; // 4 x 16B chunks per 32-elem row
      stage_chunk(&A[(size_t)(brow + row) * Kd + cc * 8],  &As[0][row * LDT + cc * 8]);
      stage_chunk(&Bm[(size_t)(bcol + row) * Kd + cc * 8], &Bs[0][row * LDT + cc * 8]);
    }
  }

  for (int t = 0; t < T; ++t) {
    stage_fence();      // our async fills have landed in LDS
    __syncthreads();    // everyone's fills visible; previous buffer reads retired
    const int cur = t & 1;

    if (t + 1 < T) {
      const int nxt = cur ^ 1;
      const int k0 = (t + 1) * BK;
#pragma unroll
      for (int s = 0; s < 2; ++s) {
        int c = tid * 2 + s;
        int row = c >> 2, cc = c & 3;
        stage_chunk(&A[(size_t)(brow + row) * Kd + k0 + cc * 8],
                    &As[nxt][row * LDT + cc * 8]);
        stage_chunk(&Bm[(size_t)(bcol + row) * Kd + k0 + cc * 8],
                    &Bs[nxt][row * LDT + cc * 8]);
        if (t + 2 < T) {
          __builtin_prefetch(&A[(size_t)(brow + row) * Kd + (t + 2) * BK + cc * 8], 0, 0);
          __builtin_prefetch(&Bm[(size_t)(bcol + row) * Kd + (t + 2) * BK + cc * 8], 0, 0);
        }
      }
    }

    // fragment fetch from LDS (matches ISA 16-bit A/B VGPR layout:
    // lane<16 -> K[0..7],K[16..23]; lane>=16 -> K[8..15],K[24..31])
    Frag a[4], b[2];
#pragma unroll
    for (int mf = 0; mf < 4; ++mf) {
      int row = wm * 64 + mf * 16 + l16;
      a[mf].q[0] = *(const uint4*)&As[cur][row * LDT + half * 8];
      a[mf].q[1] = *(const uint4*)&As[cur][row * LDT + 16 + half * 8];
    }
#pragma unroll
    for (int nf = 0; nf < 2; ++nf) {
      int row = wn * 32 + nf * 16 + l16;
      b[nf].q[0] = *(const uint4*)&Bs[cur][row * LDT + half * 8];
      b[nf].q[1] = *(const uint4*)&Bs[cur][row * LDT + 16 + half * 8];
    }
#pragma unroll
    for (int mf = 0; mf < 4; ++mf)
#pragma unroll
      for (int nf = 0; nf < 2; ++nf)
        acc[mf][nf] = __builtin_amdgcn_wmma_f32_16x16x32_bf16(
            false, a[mf].v, false, b[nf].v, (short)0, acc[mf][nf], false, false);
  }

  // epilogue: C/D layout -> VGPR r: lanes 0-15 are M=r, lanes 16-31 are M=8+r, N=lane%16
#pragma unroll
  for (int nf = 0; nf < 2; ++nf) {
    int gn = bcol + wn * 32 + nf * 16 + l16;
    float bb = bias[gn];
#pragma unroll
    for (int mf = 0; mf < 4; ++mf) {
      int gm = brow + wm * 64 + mf * 16 + half * 8;
      float* Cp = C + (size_t)gm * Ndim + gn;
#pragma unroll
      for (int r = 0; r < 8; ++r)
        Cp[(size_t)r * Ndim] = acc[mf][nf][r] + bb;
    }
  }
}

// ---------------- per-(m,h) inverse norms for q and k ----------------
__global__ __launch_bounds__(256)
void qk_norms(const float* __restrict__ qkv, float* __restrict__ qinv,
              float* __restrict__ kinv) {
  const int lane = threadIdx.x & 31;
  const int wid  = threadIdx.x >> 5;
  int id   = blockIdx.x * 8 + wid;   // 0 .. 2*B*SEQ*HH-1  (B*SEQ*HH = 65536)
  int type = id >> 16;               // 0 = q, 1 = k
  int r    = id & 65535;
  int m    = r >> 4;
  int h    = r & 15;
  const float* p = qkv + (size_t)m * (3 * DMODEL) + type * DMODEL + h * DH;
  float f0 = p[2 * lane], f1 = p[2 * lane + 1];
  float ss = f0 * f0 + f1 * f1;
#pragma unroll
  for (int msk = 16; msk >= 1; msk >>= 1) ss += __shfl_xor(ss, msk, 32);
  float inv = 1.0f / (sqrtf(ss) + 1e-6f);
  if (lane == 0) (type ? kinv : qinv)[m * HH + h] = inv;
}

// ---------------- dilated attention: one wave32 per (b,h,i) query ----------------
__global__ __launch_bounds__(256)
void dilated_attn(const float* __restrict__ qkv, const float* __restrict__ qinv,
                  const float* __restrict__ kinv, unsigned short* __restrict__ aout) {
  const int lane = threadIdx.x & 31;
  const int wid  = threadIdx.x >> 5;
  int qid  = blockIdx.x * 8 + wid;     // over B*HH*SEQ
  int i    = qid % SEQ;
  int rest = qid / SEQ;
  int h    = rest % HH;
  int b    = rest / HH;
  int m    = b * SEQ + i;

  const float* qp = qkv + (size_t)m * (3 * DMODEL) + h * DH;
  float qi = qinv[m * HH + h];
  float q0 = qp[2 * lane] * qi, q1 = qp[2 * lane + 1] * qi;

  float s[WIN];
#pragma unroll
  for (int t = 0; t < WIN; ++t) {
    int j = i - 16 + 2 * t;
    float d = -3.0e38f;
    if (j >= 0 && j < SEQ) {
      int mj = b * SEQ + j;
      const float* kp = qkv + (size_t)mj * (3 * DMODEL) + DMODEL + h * DH;
      float p = q0 * kp[2 * lane] + q1 * kp[2 * lane + 1];
#pragma unroll
      for (int msk = 16; msk >= 1; msk >>= 1) p += __shfl_xor(p, msk, 32);
      d = p * kinv[mj * HH + h];
    }
    s[t] = d;
  }
  float mx = -3.0e38f;
#pragma unroll
  for (int t = 0; t < WIN; ++t) mx = fmaxf(mx, s[t]);
  float sum = 0.f;
#pragma unroll
  for (int t = 0; t < WIN; ++t) { s[t] = __expf(s[t] - mx); sum += s[t]; }
  float inv = 1.0f / sum;

  float o0 = 0.f, o1 = 0.f;
#pragma unroll
  for (int t = 0; t < WIN; ++t) {
    int j = i - 16 + 2 * t;
    if (j >= 0 && j < SEQ) {
      int mj = b * SEQ + j;
      const float* vp = qkv + (size_t)mj * (3 * DMODEL) + 2 * DMODEL + h * DH;
      float w = s[t] * inv;
      o0 += w * vp[2 * lane];
      o1 += w * vp[2 * lane + 1];
    }
  }
  size_t o = (size_t)m * DMODEL + h * DH + 2 * lane;
  aout[o]     = f2bf(o0);
  aout[o + 1] = f2bf(o1);
}

// ---------------- launch ----------------
extern "C" void kernel_launch(void* const* d_in, const int* in_sizes, int n_in,
                              void* d_out, int out_size, void* d_ws, size_t ws_size,
                              hipStream_t stream) {
  const float* x     = (const float*)d_in[0];  // [2,2048,1024]
  const float* w_qkv = (const float*)d_in[1];  // [3072,1024]
  const float* b_qkv = (const float*)d_in[2];  // [3072]
  const float* w_out = (const float*)d_in[3];  // [1024,1024]
  const float* b_out = (const float*)d_in[4];  // [1024]
  float* out = (float*)d_out;                  // [2,2048,1024] fp32

  const int M  = BATCH * SEQ;      // 4096
  const int N1 = 3 * DMODEL;       // 3072
  const int N2 = DMODEL;           // 1024
  const int Kd = DMODEL;           // 1024

  char* w = (char*)d_ws;
  unsigned short* xb    = (unsigned short*)(w);                    // 8 MB
  unsigned short* wqkvb = (unsigned short*)(w + (8u << 20));       // 6 MB
  unsigned short* woutb = (unsigned short*)(w + (14u << 20));      // 2 MB
  float*          qkv   = (float*)(w + (16u << 20));               // 48 MB
  float*          qinv  = (float*)(w + (64u << 20));               // 256 KB
  float*          kinv  = (float*)(w + (64u << 20) + (256u << 10));// 256 KB
  unsigned short* attnb = (unsigned short*)(w + (65u << 20));      // 8 MB

  // 1) fp32 -> bf16 conversions
  {
    int n4 = (M * Kd) / 4;
    cvt_f32_bf16<<<(n4 + 255) / 256, 256, 0, stream>>>(x, xb, n4);
    n4 = (N1 * Kd) / 4;
    cvt_f32_bf16<<<(n4 + 255) / 256, 256, 0, stream>>>(w_qkv, wqkvb, n4);
    n4 = (N2 * Kd) / 4;
    cvt_f32_bf16<<<(n4 + 255) / 256, 256, 0, stream>>>(w_out, woutb, n4);
  }

  // 2) QKV projection: qkv = x @ w_qkv^T + b_qkv  (WMMA bf16, fp32 acc)
  gemm_nt_bias<<<dim3(N1 / BN, M / BM), 256, 0, stream>>>(xb, wqkvb, b_qkv, qkv, N1, Kd);

  // 3) q/k inverse norms
  qk_norms<<<(2 * M * HH) / 8, 256, 0, stream>>>(qkv, qinv, kinv);

  // 4) dilated masked attention -> bf16 [M, D]
  dilated_attn<<<(M * HH) / 8, 256, 0, stream>>>(qkv, qinv, kinv, attnb);

  // 5) output projection: out = attn @ w_out^T + b_out
  gemm_nt_bias<<<dim3(N2 / BN, M / BM), 256, 0, stream>>>(attnb, woutb, b_out, out, N2, Kd);
}
